// PersistenceViT_29841432773260
// MI455X (gfx1250) — matrix operand order repaired and checked by reading, not compiled, not run on hardware
//
#include <hip/hip_runtime.h>
#include <math.h>

// ---------------------------------------------------------------------------
// PersistenceViT forward for gfx1250 (MI455X).
//  - bf16 WMMA GEMM: 128x64 block, 8 waves, 4 WMMAs/wave/K-step, ping-pong
//    double-buffered LDS, load-all-then-store staging (loads stay in flight).
//  - Fused flash attention: 32-key tiles, full-K P*V WMMAs, next-tile
//    prefetch (global_prefetch_b8), batched fragment loads.
// ---------------------------------------------------------------------------

typedef __bf16 bf16_t;
typedef bf16_t v16bf __attribute__((ext_vector_type(16)));
typedef bf16_t v4bf  __attribute__((ext_vector_type(4)));
typedef float  v8f   __attribute__((ext_vector_type(8)));

#define GBM 128
#define GBN 64
#define GBK 32

__device__ inline void cvt4(v16bf& dst, int base, float4 f) {
    dst[base + 0] = (bf16_t)f.x; dst[base + 1] = (bf16_t)f.y;
    dst[base + 2] = (bf16_t)f.z; dst[base + 3] = (bf16_t)f.w;
}
__device__ inline v4bf pack4(float4 f) {
    v4bf h; h[0] = (bf16_t)f.x; h[1] = (bf16_t)f.y; h[2] = (bf16_t)f.z; h[3] = (bf16_t)f.w;
    return h;
}

// C[M,N] = act( A[M,K] @ W[N,K]^T + bias ) + residual   (mode 1 = exact GELU)
__global__ __launch_bounds__(256)
void gemm_wmma_kernel(const float* __restrict__ A, const float* __restrict__ W,
                      const float* __restrict__ bias, const float* __restrict__ residual,
                      float* __restrict__ C, int M, int N, int K, int mode)
{
    __shared__ bf16_t As[2][GBM][GBK];
    __shared__ bf16_t Bs[2][GBN][GBK];

    const int tid  = threadIdx.x;
    const int lane = tid & 31;
    const int wid  = tid >> 5;
    const int half = lane >> 4;
    const int l16  = lane & 15;
    const int wm   = wid & 3;     // 4 M sub-blocks of 32 rows
    const int wn   = wid >> 2;    // 2 N sub-blocks of 32 cols
    const int blockM = blockIdx.y * GBM;
    const int blockN = blockIdx.x * GBN;

    // Per-thread staging coordinates (constant across K-steps).
    int arow[4], bcol[2];
    const int skk = (tid & 7) * 4;      // float4 K offset within the 32-wide tile
#pragma unroll
    for (int i = 0; i < 4; ++i) {
        int r = (tid + i * 256) >> 3;
        int gr = blockM + r; if (gr >= M) gr = M - 1;   // clamp: never stored
        arow[i] = gr;
    }
#pragma unroll
    for (int i = 0; i < 2; ++i) {
        int n = (tid + i * 256) >> 3;
        int gn = blockN + n; if (gn >= N) gn = N - 1;
        bcol[i] = gn;
    }

    // Load-all / store-all staging so all 6 b128 loads are in flight at once.
    auto stage = [&](int buf, int k0) {
        float4 fa[4], fw[2];
#pragma unroll
        for (int i = 0; i < 4; ++i)
            fa[i] = *(const float4*)&A[(size_t)arow[i] * K + k0 + skk];
#pragma unroll
        for (int i = 0; i < 2; ++i)
            fw[i] = *(const float4*)&W[(size_t)bcol[i] * K + k0 + skk];
#pragma unroll
        for (int i = 0; i < 4; ++i)
            *(v4bf*)&As[buf][(tid + i * 256) >> 3][skk] = pack4(fa[i]);
#pragma unroll
        for (int i = 0; i < 2; ++i)
            *(v4bf*)&Bs[buf][(tid + i * 256) >> 3][skk] = pack4(fw[i]);
    };

    v8f acc00 = {}, acc01 = {}, acc10 = {}, acc11 = {};

    const int nk = K / GBK;       // all K here are multiples of 32
    stage(0, 0);
    for (int kt = 0; kt < nk; ++kt) {
        __syncthreads();          // buffer kt fully staged
        if (kt + 1 < nk) stage((kt + 1) & 1, (kt + 1) * GBK);
        const int cb = kt & 1;

        v16bf af0, af1;  // A-frags: lanes 0-15 K{0..7,16..23}, 16-31 K{8..15,24..31}
        {
            const bf16_t* a0 = &As[cb][wm * 32 + l16][0];
            const bf16_t* a1 = &As[cb][wm * 32 + 16 + l16][0];
#pragma unroll
            for (int e = 0; e < 8; ++e) {
                af0[e] = a0[half * 8 + e];  af0[e + 8] = a0[16 + half * 8 + e];
                af1[e] = a1[half * 8 + e];  af1[e + 8] = a1[16 + half * 8 + e];
            }
        }
        v16bf bf0, bf1;  // B-frags: lane = column; lanes 0-15 K{0..7,16..23}, 16-31 K{8..15,24..31}
        {
            const bf16_t* b0 = &Bs[cb][wn * 32 + l16][0];
            const bf16_t* b1 = &Bs[cb][wn * 32 + 16 + l16][0];
#pragma unroll
            for (int e = 0; e < 16; ++e) {
                bf0[e] = b0[(e >> 3) * 16 + half * 8 + (e & 7)];
                bf1[e] = b1[(e >> 3) * 16 + half * 8 + (e & 7)];
            }
        }
        acc00 = __builtin_amdgcn_wmma_f32_16x16x32_bf16(false, af0, false, bf0, (short)0, acc00, false, false);
        acc01 = __builtin_amdgcn_wmma_f32_16x16x32_bf16(false, af0, false, bf1, (short)0, acc01, false, false);
        acc10 = __builtin_amdgcn_wmma_f32_16x16x32_bf16(false, af1, false, bf0, (short)0, acc10, false, false);
        acc11 = __builtin_amdgcn_wmma_f32_16x16x32_bf16(false, af1, false, bf1, (short)0, acc11, false, false);
    }

    const int col0 = blockN + wn * 32 + l16;
    const int col1 = col0 + 16;
#pragma unroll
    for (int mt = 0; mt < 2; ++mt) {
        v8f* aL = mt ? &acc10 : &acc00;
        v8f* aR = mt ? &acc11 : &acc01;
#pragma unroll
        for (int r = 0; r < 8; ++r) {
            int row = blockM + wm * 32 + mt * 16 + r + half * 8;
            if (row >= M) continue;
            size_t rb = (size_t)row * N;
            if (col0 < N) {
                float f = (*aL)[r] + (bias ? bias[col0] : 0.f);
                if (mode == 1) f = 0.5f * f * (1.f + erff(f * 0.70710678f));
                if (residual) f += residual[rb + col0];
                C[rb + col0] = f;
            }
            if (col1 < N) {
                float f = (*aR)[r] + (bias ? bias[col1] : 0.f);
                if (mode == 1) f = 0.5f * f * (1.f + erff(f * 0.70710678f));
                if (residual) f += residual[rb + col1];
                C[rb + col1] = f;
            }
        }
    }
}

// ---------------------------------------------------------------------------
// Flash attention, one wave per (b, head, 16-query tile), 32 keys/iteration.
// qkv: [B*580, 2304], col = i*768 + head*64 + d. pbias/importance logit
// offsets are constant along the softmax axis -> cancel -> omitted.
// ---------------------------------------------------------------------------
__global__ __launch_bounds__(32)
void attn_flash_kernel(const float* __restrict__ qkv, float* __restrict__ o)
{
    const int qt   = blockIdx.x;   // 0..36
    const int hh   = blockIdx.y;   // 0..11
    const int b    = blockIdx.z;   // 0..31
    const int lane = threadIdx.x;
    const int half = lane >> 4;
    const int l16  = lane & 15;
    const int qbase = qt * 16;
    const float scale = 0.125f;    // 64^-0.5

    __shared__ bf16_t Plds[16][32];     // P tile, C-order -> A-order bounce
    __shared__ bf16_t VtT[64][40];      // V transposed: [d][key], padded rows

    int qrow = qbase + l16; if (qrow > 579) qrow = 579;
    const float* qptr = qkv + ((size_t)(b * 580 + qrow)) * 2304 + hh * 64;
    v16bf qf0, qf1;    // A-frag mapping: e<8 -> d=half*8+e ; e>=8 -> d=16+half*8+(e-8)
    cvt4(qf0, 0,  *(const float4*)(qptr + half * 8));
    cvt4(qf0, 4,  *(const float4*)(qptr + half * 8 + 4));
    cvt4(qf0, 8,  *(const float4*)(qptr + 16 + half * 8));
    cvt4(qf0, 12, *(const float4*)(qptr + 16 + half * 8 + 4));
    cvt4(qf1, 0,  *(const float4*)(qptr + 32 + half * 8));
    cvt4(qf1, 4,  *(const float4*)(qptr + 32 + half * 8 + 4));
    cvt4(qf1, 8,  *(const float4*)(qptr + 48 + half * 8));
    cvt4(qf1, 12, *(const float4*)(qptr + 48 + half * 8 + 4));

    v8f o0 = {}, o1 = {}, o2 = {}, o3 = {};
    float mi[8], li[8];
#pragma unroll
    for (int r = 0; r < 8; ++r) { mi[r] = -3.0e38f; li[r] = 0.f; }

    for (int kt = 0; kt < 19; ++kt) {            // ceil(580/32)
        const int kbase = kt * 32;
        __syncthreads();                          // prior-iter LDS reads done

        // Prefetch next key tile's K and V rows into cache (global_prefetch_b8).
        if (kt + 1 < 19) {
            int pr = kbase + 32 + lane; if (pr > 579) pr = 579;
            const float* pk = qkv + ((size_t)(b * 580 + pr)) * 2304 + 768 + hh * 64;
            __builtin_prefetch(pk, 0, 1);
            __builtin_prefetch(pk + 768, 0, 1);   // matching V row
        }

        // Stage V (32 keys x 64 dims): all 16 b128 loads in flight, then the
        // transposed bf16 LDS stores.
        int vrow = kbase + lane; if (vrow > 579) vrow = 579;
        const float* vptr = qkv + ((size_t)(b * 580 + vrow)) * 2304 + 1536 + hh * 64;
        {
            float4 fv[16];
#pragma unroll
            for (int q4 = 0; q4 < 16; ++q4) fv[q4] = *(const float4*)(vptr + q4 * 4);
#pragma unroll
            for (int q4 = 0; q4 < 16; ++q4) {
                VtT[q4 * 4 + 0][lane] = (bf16_t)fv[q4].x;
                VtT[q4 * 4 + 1][lane] = (bf16_t)fv[q4].y;
                VtT[q4 * 4 + 2][lane] = (bf16_t)fv[q4].z;
                VtT[q4 * 4 + 3][lane] = (bf16_t)fv[q4].w;
            }
        }

        // Two 16-key score tiles: batch all 16 K-row loads, then 4 WMMAs.
        int k0r = kbase + l16;      bool kv0 = k0r < 580; if (!kv0) k0r = 579;
        int k1r = kbase + 16 + l16; bool kv1 = k1r < 580; if (!kv1) k1r = 579;
        const float* kp0 = qkv + ((size_t)(b * 580 + k0r)) * 2304 + 768 + hh * 64;
        const float* kp1 = qkv + ((size_t)(b * 580 + k1r)) * 2304 + 768 + hh * 64;
        v8f S0 = {}, S1 = {};
        {
            float4 ff[16];
#pragma unroll
            for (int i = 0; i < 4; ++i) ff[i]      = *(const float4*)(kp0 + (i >> 1) * 16 + half * 8 + (i & 1) * 4);
#pragma unroll
            for (int i = 0; i < 4; ++i) ff[4 + i]  = *(const float4*)(kp0 + 32 + (i >> 1) * 16 + half * 8 + (i & 1) * 4);
#pragma unroll
            for (int i = 0; i < 4; ++i) ff[8 + i]  = *(const float4*)(kp1 + (i >> 1) * 16 + half * 8 + (i & 1) * 4);
#pragma unroll
            for (int i = 0; i < 4; ++i) ff[12 + i] = *(const float4*)(kp1 + 32 + (i >> 1) * 16 + half * 8 + (i & 1) * 4);
            v16bf kf00, kf01, kf10, kf11;   // B-frags: lane = key col
#pragma unroll
            for (int i = 0; i < 4; ++i) {
                cvt4(kf00, i * 4, ff[i]);
                cvt4(kf01, i * 4, ff[4 + i]);
                cvt4(kf10, i * 4, ff[8 + i]);
                cvt4(kf11, i * 4, ff[12 + i]);
            }
            S0 = __builtin_amdgcn_wmma_f32_16x16x32_bf16(false, qf0, false, kf00, (short)0, S0, false, false);
            S0 = __builtin_amdgcn_wmma_f32_16x16x32_bf16(false, qf1, false, kf01, (short)0, S0, false, false);
            S1 = __builtin_amdgcn_wmma_f32_16x16x32_bf16(false, qf0, false, kf10, (short)0, S1, false, false);
            S1 = __builtin_amdgcn_wmma_f32_16x16x32_bf16(false, qf1, false, kf11, (short)0, S1, false, false);
        }

        // Online softmax across the 32 keys.
        float alpha[8], p0v[8], p1v[8];
#pragma unroll
        for (int r = 0; r < 8; ++r) {
            float s0 = kv0 ? S0[r] * scale : -3.0e38f;
            float s1 = kv1 ? S1[r] * scale : -3.0e38f;
            float mx = fmaxf(s0, s1);
            mx = fmaxf(mx, __shfl_xor(mx, 1));
            mx = fmaxf(mx, __shfl_xor(mx, 2));
            mx = fmaxf(mx, __shfl_xor(mx, 4));
            mx = fmaxf(mx, __shfl_xor(mx, 8));
            float mnew = fmaxf(mi[r], mx);
            float a  = __expf(mi[r] - mnew);
            float p0 = __expf(s0 - mnew);
            float p1 = __expf(s1 - mnew);
            float ps = p0 + p1;
            ps += __shfl_xor(ps, 1);
            ps += __shfl_xor(ps, 2);
            ps += __shfl_xor(ps, 4);
            ps += __shfl_xor(ps, 8);
            li[r] = li[r] * a + ps;
            mi[r] = mnew;
            alpha[r] = a; p0v[r] = p0; p1v[r] = p1;
        }
#pragma unroll
        for (int r = 0; r < 8; ++r) {
            Plds[r + half * 8][l16]      = (bf16_t)p0v[r];
            Plds[r + half * 8][16 + l16] = (bf16_t)p1v[r];
        }
        __syncthreads();

        // P as full 16x32 A-fragment.
        v16bf pf;
        {
            const bf16_t* prow = &Plds[l16][0];
#pragma unroll
            for (int e = 0; e < 8; ++e) {
                pf[e]     = prow[half * 8 + e];
                pf[e + 8] = prow[16 + half * 8 + e];
            }
        }
#pragma unroll
        for (int r = 0; r < 8; ++r) {
            o0[r] *= alpha[r]; o1[r] *= alpha[r];
            o2[r] *= alpha[r]; o3[r] *= alpha[r];
        }
        // V B-frags from transposed LDS: keys {half*8..+7, 16+half*8..+7} per lane.
#pragma unroll
        for (int j = 0; j < 4; ++j) {
            const bf16_t* vrow2 = &VtT[j * 16 + l16][0];
            v16bf vf;
#pragma unroll
            for (int e = 0; e < 16; ++e) vf[e] = vrow2[(e >> 3) * 16 + half * 8 + (e & 7)];
            v8f* op = (j == 0) ? &o0 : (j == 1) ? &o1 : (j == 2) ? &o2 : &o3;
            *op = __builtin_amdgcn_wmma_f32_16x16x32_bf16(false, pf, false, vf, (short)0, *op, false, false);
        }
    }

#pragma unroll
    for (int r = 0; r < 8; ++r) {
        int n = qbase + r + half * 8;
        if (n < 580) {
            float inv = 1.0f / li[r];
            size_t base = ((size_t)(b * 580 + n)) * 768 + hh * 64;
            o[base +  0 + l16] = o0[r] * inv;
            o[base + 16 + l16] = o1[r] * inv;
            o[base + 32 + l16] = o2[r] * inv;
            o[base + 48 + l16] = o3[r] * inv;
        }
    }
}

// ---------------------------------------------------------------------------
// Supporting kernels
// ---------------------------------------------------------------------------
__global__ void im2col_kernel(const float* __restrict__ img, float* __restrict__ out)
{
    int i4 = blockIdx.x * 256 + threadIdx.x;            // one float4 each
    if (i4 >= 32 * 576 * 192) return;
    int idx = i4 * 4;
    int col = idx % 768;
    int row = idx / 768;
    int b = row / 576, n = row % 576;
    int c = col >> 8, py = (col >> 4) & 15, px = col & 15;   // px multiple of 4
    int ny = n / 24, nx = n % 24;
    float4 f = *(const float4*)&img[(((size_t)b * 3 + c) * 384 + ny * 16 + py) * 384 + nx * 16 + px];
    *(float4*)&out[idx] = f;
}

__global__ __launch_bounds__(32)
void importance_kernel(const float* __restrict__ im2col, const float* __restrict__ w1,
                       const float* __restrict__ b1, const float* __restrict__ w2,
                       const float* __restrict__ b2, float* __restrict__ imp)
{
    __shared__ float ch[32];
    int p = blockIdx.x;
    int t = threadIdx.x;
    const float4* a = (const float4*)(im2col + (size_t)p * 768);
    const float4* w = (const float4*)(w1 + (size_t)t * 768);
    float acc = b1[t];
    for (int i = 0; i < 192; ++i) {
        float4 x = a[i], y = w[i];
        acc += x.x * y.x + x.y * y.y + x.z * y.z + x.w * y.w;
    }
    ch[t] = fmaxf(acc, 0.f);
    __syncthreads();
    if (t == 0) {
        float v = b2[0];
        for (int i = 0; i < 32; ++i) v += ch[i] * w2[i];
        imp[p] = 1.f / (1.f + __expf(-v));
    }
}

__global__ void assemble_kernel(const float* __restrict__ patches, const float* __restrict__ imp,
                                const float* __restrict__ cls, const float* __restrict__ hom,
                                const float* __restrict__ pos, float* __restrict__ x)
{
    int i4 = blockIdx.x * 256 + threadIdx.x;
    if (i4 >= 32 * 580 * 192) return;
    int idx = i4 * 4;
    int d = idx % 768;
    int t = (idx / 768) % 580;
    int b = idx / (768 * 580);
    float4 v;
    if (t == 0)     v = *(const float4*)&cls[d];
    else if (t < 4) v = *(const float4*)&hom[(t - 1) * 768 + d];
    else {
        int n = t - 4;
        float4 p = *(const float4*)&patches[((size_t)b * 576 + n) * 768 + d];
        float g = 1.f + imp[b * 576 + n];
        v.x = p.x * g; v.y = p.y * g; v.z = p.z * g; v.w = p.w * g;
        if (d < 384) {
            float4 q = *(const float4*)&pos[n * 384 + d];
            v.x += q.x; v.y += q.y; v.z += q.z; v.w += q.w;
        }
    }
    *(float4*)&x[idx] = v;
}

__global__ __launch_bounds__(256)
void layernorm_kernel(const float* __restrict__ x, const float* __restrict__ g,
                      const float* __restrict__ bb, float* __restrict__ y)
{
    __shared__ float red[256];
    int row = blockIdx.x, t = threadIdx.x;
    const float* xr = x + (size_t)row * 768;
    float v0 = xr[t], v1 = xr[t + 256], v2 = xr[t + 512];
    red[t] = v0 + v1 + v2;
    __syncthreads();
    for (int s = 128; s > 0; s >>= 1) { if (t < s) red[t] += red[t + s]; __syncthreads(); }
    float mean = red[0] * (1.f / 768.f);
    __syncthreads();
    float d0 = v0 - mean, d1 = v1 - mean, d2 = v2 - mean;
    red[t] = d0 * d0 + d1 * d1 + d2 * d2;
    __syncthreads();
    for (int s = 128; s > 0; s >>= 1) { if (t < s) red[t] += red[t + s]; __syncthreads(); }
    float rstd = rsqrtf(red[0] * (1.f / 768.f) + 1e-5f);
    float* yr = y + (size_t)row * 768;
    yr[t]       = d0 * rstd * g[t]       + bb[t];
    yr[t + 256] = d1 * rstd * g[t + 256] + bb[t + 256];
    yr[t + 512] = d2 * rstd * g[t + 512] + bb[t + 512];
}

// Aggregation head: per-batch cross-attention (cls -> homology), LN, weighted sum.
__global__ __launch_bounds__(256)
void agg_kernel(const float* __restrict__ h,
                const float* __restrict__ in_w, const float* __restrict__ in_b,
                const float* __restrict__ out_w, const float* __restrict__ out_b,
                const float* __restrict__ ln_g, const float* __restrict__ ln_b,
                const float* __restrict__ aw, float* __restrict__ fin)
{
    __shared__ float q[768], kk[3 * 768], vv[3 * 768], agg[768], outv[768];
    __shared__ float sc[4][3], attw[4][3], red[256];
    int b = blockIdx.x, t = threadIdx.x;
    const float* cls = h + (size_t)(b * 580) * 768;
    for (int d = t; d < 768; d += 256) {
        float accq = in_b[d];
        const float* wr = in_w + (size_t)d * 768;
        for (int i = 0; i < 768; ++i) accq += cls[i] * wr[i];
        q[d] = accq;
        for (int j = 0; j < 3; ++j) {
            const float* tok = h + (size_t)(b * 580 + 1 + j) * 768;
            const float* wkr = in_w + (size_t)(768 + d) * 768;
            const float* wvr = in_w + (size_t)(1536 + d) * 768;
            float sk = in_b[768 + d], sv = in_b[1536 + d];
            for (int i = 0; i < 768; ++i) { sk += tok[i] * wkr[i]; sv += tok[i] * wvr[i]; }
            kk[j * 768 + d] = sk; vv[j * 768 + d] = sv;
        }
    }
    __syncthreads();
    if (t < 12) {
        int h4 = t / 3, j = t % 3;
        float s = 0.f;
        for (int i = 0; i < 192; ++i) s += q[h4 * 192 + i] * kk[j * 768 + h4 * 192 + i];
        sc[h4][j] = s * 0.0721687836487032f;   // 192^-0.5
    }
    __syncthreads();
    if (t < 4) {
        float m = fmaxf(sc[t][0], fmaxf(sc[t][1], sc[t][2]));
        float e0 = __expf(sc[t][0] - m), e1 = __expf(sc[t][1] - m), e2 = __expf(sc[t][2] - m);
        float inv = 1.f / (e0 + e1 + e2);
        attw[t][0] = e0 * inv; attw[t][1] = e1 * inv; attw[t][2] = e2 * inv;
    }
    __syncthreads();
    for (int d = t; d < 768; d += 256) {
        int h4 = d / 192;
        agg[d] = attw[h4][0] * vv[d] + attw[h4][1] * vv[768 + d] + attw[h4][2] * vv[1536 + d];
    }
    __syncthreads();
    for (int d = t; d < 768; d += 256) {
        float a = out_b[d];
        const float* wr = out_w + (size_t)d * 768;
        for (int i = 0; i < 768; ++i) a += agg[i] * wr[i];
        outv[d] = a;
    }
    __syncthreads();
    red[t] = outv[t] + outv[t + 256] + outv[t + 512];
    __syncthreads();
    for (int s = 128; s > 0; s >>= 1) { if (t < s) red[t] += red[t + s]; __syncthreads(); }
    float mean = red[0] * (1.f / 768.f);
    __syncthreads();
    float d0 = outv[t] - mean, d1 = outv[t + 256] - mean, d2 = outv[t + 512] - mean;
    red[t] = d0 * d0 + d1 * d1 + d2 * d2;
    __syncthreads();
    for (int s = 128; s > 0; s >>= 1) { if (t < s) red[t] += red[t + s]; __syncthreads(); }
    float rstd = rsqrtf(red[0] * (1.f / 768.f) + 1e-5f);

    float w0 = aw[0], w1 = aw[1], w2 = aw[2], w3 = aw[3];
    float wm = fmaxf(fmaxf(w0, w1), fmaxf(w2, w3));
    float e0 = __expf(w0 - wm), e1 = __expf(w1 - wm), e2 = __expf(w2 - wm), e3 = __expf(w3 - wm);
    float wi = 1.f / (e0 + e1 + e2 + e3);
    e0 *= wi; e1 *= wi; e2 *= wi; e3 *= wi;

    const float* t0 = h + (size_t)(b * 580) * 768;
    float* fo = fin + (size_t)b * 768;
    int   dd[3] = { t, t + 256, t + 512 };
    float dv[3] = { d0, d1, d2 };
    for (int i = 0; i < 3; ++i) {
        int d = dd[i];
        float ln = dv[i] * rstd * ln_g[d] + ln_b[d];
        float sum = t0[d] * e0 + t0[768 + d] * e1 + t0[2 * 768 + d] * e2 + t0[3 * 768 + d] * e3;
        fo[d] = ln + sum;
    }
}

// ---------------------------------------------------------------------------
extern "C" void kernel_launch(void* const* d_in, const int* in_sizes, int n_in,
                              void* d_out, int out_size, void* d_ws, size_t ws_size,
                              hipStream_t stream)
{
    const float* img     = (const float*)d_in[0];
    const float* conv_w  = (const float*)d_in[1];
    const float* conv_b  = (const float*)d_in[2];
    const float* imp1_w  = (const float*)d_in[3];
    const float* imp1_b  = (const float*)d_in[4];
    const float* imp2_w  = (const float*)d_in[5];
    const float* imp2_b  = (const float*)d_in[6];
    const float* spos    = (const float*)d_in[7];
    const float* cls     = (const float*)d_in[8];
    const float* hom     = (const float*)d_in[9];
    const float* ln1_g   = (const float*)d_in[10];
    const float* ln1_b   = (const float*)d_in[11];
    const float* qkv_w   = (const float*)d_in[12];
    // d_in[13] = blk_pbias: constant along softmax axis -> cancels, unused.
    const float* proj_w  = (const float*)d_in[14];
    const float* proj_b  = (const float*)d_in[15];
    const float* ln2_g   = (const float*)d_in[16];
    const float* ln2_b   = (const float*)d_in[17];
    const float* mlp1_w  = (const float*)d_in[18];
    const float* mlp1_b  = (const float*)d_in[19];
    const float* mlp2_w  = (const float*)d_in[20];
    const float* mlp2_b  = (const float*)d_in[21];
    const float* agg_in_w  = (const float*)d_in[22];
    const float* agg_in_b  = (const float*)d_in[23];
    const float* agg_out_w = (const float*)d_in[24];
    const float* agg_out_b = (const float*)d_in[25];
    const float* agg_ln_g  = (const float*)d_in[26];
    const float* agg_ln_b  = (const float*)d_in[27];
    const float* agg_wts   = (const float*)d_in[28];
    const float* norm_g    = (const float*)d_in[29];
    const float* norm_b    = (const float*)d_in[30];
    const float* head_w    = (const float*)d_in[31];
    const float* head_b    = (const float*)d_in[32];
    float* out = (float*)d_out;

    constexpr int ROWS = 32 * 580;                 // 18560 token rows
    constexpr size_t N_X   = (size_t)ROWS * 768;   // 14,254,080
    constexpr size_t N_QKV = (size_t)ROWS * 2304;  // 42,762,240
    constexpr size_t N_I2C = (size_t)32 * 576 * 768;

    float* X   = (float*)d_ws;
    float* H   = X + N_X;
    float* QKV = H + N_X;
    float* O   = QKV + N_QKV;
    float* FIN = O + N_X;
    float* I2C   = H;            // im2col aliases H (used before LN exists)
    float* PATCH = QKV;          // patch embeddings alias QKV
    float* IMP   = QKV + N_I2C;  // importance gates (18432 floats)
    float* HID   = O;            // MLP hidden chunk aliases O (4640*3072 == N_X)

    auto gemm = [&](const float* A, const float* W, const float* bias, const float* res,
                    float* C, int M, int N, int K, int mode) {
        dim3 g((N + GBN - 1) / GBN, (M + GBM - 1) / GBM);
        gemm_wmma_kernel<<<g, 256, 0, stream>>>(A, W, bias, res, C, M, N, K, mode);
    };

    // 1) patch embed = im2col + GEMM against conv_w (768x768, matching layout)
    im2col_kernel<<<(int)((N_I2C / 4 + 255) / 256), 256, 0, stream>>>(img, I2C);
    gemm(I2C, conv_w, conv_b, nullptr, PATCH, 32 * 576, 768, 768, 0);

    // 2) importance gate (reuses im2col; imp1_w flattens identically)
    importance_kernel<<<32 * 576, 32, 0, stream>>>(I2C, imp1_w, imp1_b, imp2_w, imp2_b, IMP);

    // 3) token assembly: cls | homology | gated patches + half positional enc
    assemble_kernel<<<(int)((N_X / 4 + 255) / 256), 256, 0, stream>>>(PATCH, IMP, cls, hom, spos, X);

    // 4) transformer blocks
    for (int l = 0; l < 12; ++l) {
        layernorm_kernel<<<ROWS, 256, 0, stream>>>(X, ln1_g + l * 768, ln1_b + l * 768, H);
        gemm(H, qkv_w + (size_t)l * 2304 * 768, nullptr, nullptr, QKV, ROWS, 2304, 768, 0);
        attn_flash_kernel<<<dim3(37, 12, 32), 32, 0, stream>>>(QKV, O);
        gemm(O, proj_w + (size_t)l * 768 * 768, proj_b + l * 768, X, X, ROWS, 768, 768, 0);
        layernorm_kernel<<<ROWS, 256, 0, stream>>>(X, ln2_g + l * 768, ln2_b + l * 768, H);
        for (int c = 0; c < 4; ++c) {       // chunk MLP rows to bound scratch
            int r0 = c * 4640;
            gemm(H + (size_t)r0 * 768, mlp1_w + (size_t)l * 3072 * 768, mlp1_b + l * 3072,
                 nullptr, HID, 4640, 3072, 768, 1 /*GELU*/);
            gemm(HID, mlp2_w + (size_t)l * 768 * 3072, mlp2_b + l * 768,
                 X + (size_t)r0 * 768, X + (size_t)r0 * 768, 4640, 768, 3072, 0);
        }
    }

    // 5) final LN, aggregation head, classifier
    layernorm_kernel<<<ROWS, 256, 0, stream>>>(X, norm_g, norm_b, H);
    agg_kernel<<<32, 256, 0, stream>>>(H, agg_in_w, agg_in_b, agg_out_w, agg_out_b,
                                       agg_ln_g, agg_ln_b, agg_wts, FIN);
    gemm(FIN, head_w, head_b, nullptr, out, 32, 1000, 768, 0);

    (void)in_sizes; (void)n_in; (void)out_size; (void)ws_size;
}
